// MultiHeadAttention_45681272160492
// MI455X (gfx1250) — compile-verified, matrix operand order checked by reading, not run on hardware
//
#include <hip/hip_runtime.h>

#define DEV_INLINE __device__ __forceinline__

typedef __attribute__((ext_vector_type(16))) __bf16 v16bf;
typedef __attribute__((ext_vector_type(8)))  float  v8f;

constexpr int Bd = 2;
constexpr int S  = 2048;
constexpr int D  = 1024;
constexpr int H  = 16;
constexpr int DK = 64;

union FragBf { v16bf v; __bf16 e[16]; };
union AccF   { v8f  v; float  f[8]; };

// ---- async global->LDS copy (CDNA5 GLOBAL_LOAD_ASYNC_TO_LDS_B128) ----------
#if defined(__has_builtin)
#if __has_builtin(__builtin_amdgcn_global_load_async_to_lds_b128) && \
    __has_builtin(__builtin_amdgcn_s_wait_asynccnt)
#define USE_ASYNC_COPY 1
#endif
#endif
#ifndef USE_ASYNC_COPY
#define USE_ASYNC_COPY 0
#endif

typedef int b128_t __attribute__((vector_size(16)));
typedef __attribute__((address_space(1))) b128_t as1_b128;
typedef __attribute__((address_space(3))) b128_t as3_b128;

DEV_INLINE void copy_b128_to_lds(void* lds_dst, const void* gsrc) {
#if USE_ASYNC_COPY
    __builtin_amdgcn_global_load_async_to_lds_b128((as1_b128*)gsrc,
                                                   (as3_b128*)lds_dst, 0, 0);
#else
    *(uint4*)lds_dst = *(const uint4*)gsrc;
#endif
}

DEV_INLINE void copy_fence() {
#if USE_ASYNC_COPY
    __builtin_amdgcn_s_wait_asynccnt(0);
#endif
}

// ---- bf16 pair packing (lowers to v_cvt_pk_bf16_f32) ------------------------
DEV_INLINE unsigned pk_bf16(float a, float b) {
    union { __bf16 h[2]; unsigned u; } p;
    p.h[0] = (__bf16)a;
    p.h[1] = (__bf16)b;
    return p.u;
}

DEV_INLINE unsigned pk_u16(unsigned short a, unsigned short b) {
    return (unsigned)a | ((unsigned)b << 16);
}

// K-index pattern for 16-bit A/B fragments (ISA 7.12.2, 16-bit A 16x32):
// VGPR v (0..3): K = half*8 + 2v {+0,+1};  VGPR v (4..7): K = 16 + half*8 + 2(v-4)
DEV_INLINE int frag_k(int v, int half) {
    return (v < 4) ? (half * 8 + 2 * v) : (16 + half * 8 + 2 * (v - 4));
}

// Load one 16x32 bf16 operand fragment from LDS laid out [row][k], row stride
// `ld` elements.  A: row = M.  B: stored [N][K] so the same loader applies.
DEV_INLINE v16bf load_frag_mk(const __bf16* base, int ld, int lane) {
    FragBf f;
    const int r = lane & 15;
    const int h = lane >> 4;
    const __bf16* src = base + r * ld;
#pragma unroll
    for (int v = 0; v < 8; ++v) {
        const int k0 = frag_k(v, h);
        f.e[2 * v]     = src[k0];
        f.e[2 * v + 1] = src[k0 + 1];
    }
    return f.v;
}

DEV_INLINE v8f wmma_bf16(v16bf a, v16bf b, v8f c) {
    return __builtin_amdgcn_wmma_f32_16x16x32_bf16(false, a, false, b, (short)0, c,
                                                   false, false);
}

// ---------------------------------------------------------------------------
// Kernel 1: QKV projections.  Y = X*W + bias -> bf16 head-split layout
// [(b*H+h)*S + s][dk].  Grid: (M/64, D/64, 3); block 256 (8 waves).
// K-chunk = 64: lA[64x64] (m-major), lB[64x64] ([n][k]).
// ---------------------------------------------------------------------------
__global__ __launch_bounds__(256) void qkv_proj_kernel(
    const float* __restrict__ q, const float* __restrict__ k,
    const float* __restrict__ v, const float* __restrict__ Wq,
    const float* __restrict__ bq, const float* __restrict__ Wk,
    const float* __restrict__ bk, const float* __restrict__ Wv,
    const float* __restrict__ bv, __bf16* __restrict__ Qb,
    __bf16* __restrict__ Kb, __bf16* __restrict__ Vb) {
    const int z = blockIdx.z;
    const float* X    = (z == 0) ? q  : (z == 1) ? k  : v;
    const float* W    = (z == 0) ? Wq : (z == 1) ? Wk : Wv;
    const float* bias = (z == 0) ? bq : (z == 1) ? bk : bv;
    __bf16* O         = (z == 0) ? Qb : (z == 1) ? Kb : Vb;

    __shared__ __bf16 lA[64 * 64];  // X tile  [m][k]
    __shared__ __bf16 lB[64 * 64];  // W tile transposed: [n][k]

    const int tid  = threadIdx.x;
    const int lane = tid & 31;
    const int wave = tid >> 5;
    const int m0   = blockIdx.x * 64;
    const int n0   = blockIdx.y * 64;

    AccF acc[2];
#pragma unroll
    for (int t = 0; t < 2; ++t) acc[t].v = (v8f){0, 0, 0, 0, 0, 0, 0, 0};

    for (int k0 = 0; k0 < D; k0 += 64) {
        // A: 64x64 fp32 -> bf16, float4 loads + b64 LDS stores
#pragma unroll
        for (int i = 0; i < 4; ++i) {
            const int idx = tid + i * 256;          // 1024 float4s
            const int r = idx >> 4, c4 = idx & 15;  // 16 float4 per row
            const float4 xv =
                ((const float4*)(X + (size_t)(m0 + r) * D + k0))[c4];
            uint2 pv;
            pv.x = pk_bf16(xv.x, xv.y);
            pv.y = pk_bf16(xv.z, xv.w);
            ((uint2*)lA)[r * 16 + c4] = pv;
        }
        // B: 64x64 W chunk transposed into [n][k]; pack k-pairs, b32 stores
#pragma unroll
        for (int i = 0; i < 8; ++i) {
            const int idx = tid + i * 256;          // 64 nn x 32 k-pairs
            const int kp = idx >> 6, nn = idx & 63;
            const size_t g0 = (size_t)(k0 + 2 * kp) * D + (n0 + nn);
            ((unsigned*)lB)[nn * 32 + kp] = pk_bf16(W[g0], W[g0 + D]);
            if (k0 + 64 < D)
                __builtin_prefetch(&W[g0 + (size_t)64 * D], 0, 1);
        }
        __syncthreads();
#pragma unroll
        for (int t = 0; t < 2; ++t) {
            const int tile = wave * 2 + t;  // 16 tiles over 8 waves
            const int qt = tile >> 2, nt = tile & 3;
#pragma unroll
            for (int kk = 0; kk < 64; kk += 32) {
                const v16bf fa = load_frag_mk(lA + qt * 16 * 64 + kk, 64, lane);
                const v16bf fb = load_frag_mk(lB + nt * 16 * 64 + kk, 64, lane);
                acc[t].v = wmma_bf16(fa, fb, acc[t].v);
            }
        }
        __syncthreads();
    }

#pragma unroll
    for (int t = 0; t < 2; ++t) {
        const int tile = wave * 2 + t;
        const int qt = tile >> 2, nt = tile & 3;
        const int nl = lane & 15, hh = lane >> 4;
        const int n  = n0 + nt * 16 + nl;
        const int h  = n >> 6, dn = n & 63;
        const float bb = bias[n];
#pragma unroll
        for (int i = 0; i < 8; ++i) {
            const int m = m0 + qt * 16 + i + hh * 8;
            const int b = m >> 11;  // m / S
            const int s = m & (S - 1);
            O[(size_t)((b * H + h) * S + s) * DK + dn] = (__bf16)(acc[t].f[i] + bb);
        }
    }
}

// ---------------------------------------------------------------------------
// Kernel 2: scores = scale * Q K^T with key-padding mask -> attn (pre-softmax).
// Grid: (S/64 q-tiles, S/64 key-tiles, B*H); block 256.  Q/K tiles staged via
// async global->LDS b128 copies when available.
// ---------------------------------------------------------------------------
__global__ __launch_bounds__(256) void scores_kernel(
    const __bf16* __restrict__ Qb, const __bf16* __restrict__ Kb,
    const int* __restrict__ mask, float* __restrict__ attn) {
    const int bh = blockIdx.z;
    const int b  = bh / H;
    const int q0 = blockIdx.x * 64;
    const int p0 = blockIdx.y * 64;

    __shared__ __bf16 lQ[64 * 64];  // [q][dk]
    __shared__ __bf16 lK[64 * 64];  // [key][dk]  (== B as [n][k])

    const int tid = threadIdx.x, lane = tid & 31, wave = tid >> 5;
    const __bf16* Qh = Qb + (size_t)bh * S * DK;
    const __bf16* Kh = Kb + (size_t)bh * S * DK;

    // 512 b128 transfers per tile (8 uint4 per 64xDK row)
#pragma unroll
    for (int i = 0; i < 2; ++i) {
        const int idx = tid + i * 256;
        const int r = idx >> 3, c = idx & 7;
        copy_b128_to_lds(&((uint4*)lQ)[idx],
                         (const uint4*)(Qh + (size_t)(q0 + r) * DK) + c);
        copy_b128_to_lds(&((uint4*)lK)[idx],
                         (const uint4*)(Kh + (size_t)(p0 + r) * DK) + c);
    }
    copy_fence();
    __syncthreads();

    const float scale = 0.125f;  // 1/sqrt(64)
#pragma unroll
    for (int t = 0; t < 2; ++t) {
        const int tile = wave * 2 + t;
        const int qt = tile >> 2, pt = tile & 3;
        AccF acc;
        acc.v = (v8f){0, 0, 0, 0, 0, 0, 0, 0};
#pragma unroll
        for (int kk = 0; kk < DK; kk += 32) {
            const v16bf fa = load_frag_mk(lQ + qt * 16 * 64 + kk, 64, lane);
            const v16bf fb = load_frag_mk(lK + pt * 16 * 64 + kk, 64, lane);
            acc.v = wmma_bf16(fa, fb, acc.v);
        }
        const int nl = lane & 15, hh = lane >> 4;
        const int p  = p0 + pt * 16 + nl;
        const bool live = mask[b * S + p] != 0;
#pragma unroll
        for (int i = 0; i < 8; ++i) {
            const int qq = q0 + qt * 16 + i + hh * 8;
            const float sc = live ? acc.f[i] * scale : -1.0e30f;
            attn[(size_t)bh * S * S + (size_t)qq * S + p] = sc;
        }
    }
}

// ---------------------------------------------------------------------------
// Kernel 3: in-place row softmax over attn rows of length S=2048.
// Grid: B*H*S blocks of 256 threads (two float4 per thread).
// ---------------------------------------------------------------------------
__global__ __launch_bounds__(256) void softmax_kernel(float* __restrict__ attn) {
    float4* p4 = (float4*)(attn + (size_t)blockIdx.x * S);
    const int tid = threadIdx.x;

    __shared__ float redMax[8];
    __shared__ float redSum[8];

    float4 x[2];
    float mx = -3.0e38f;
#pragma unroll
    for (int i = 0; i < 2; ++i) {
        x[i] = p4[tid + i * 256];
        mx = fmaxf(mx, fmaxf(fmaxf(x[i].x, x[i].y), fmaxf(x[i].z, x[i].w)));
    }
#pragma unroll
    for (int off = 16; off > 0; off >>= 1) mx = fmaxf(mx, __shfl_xor(mx, off, 32));
    if ((tid & 31) == 0) redMax[tid >> 5] = mx;
    __syncthreads();
    float m = redMax[0];
#pragma unroll
    for (int w = 1; w < 8; ++w) m = fmaxf(m, redMax[w]);

    float sum = 0.0f;
#pragma unroll
    for (int i = 0; i < 2; ++i) {
        x[i].x = __expf(x[i].x - m);
        x[i].y = __expf(x[i].y - m);
        x[i].z = __expf(x[i].z - m);
        x[i].w = __expf(x[i].w - m);
        sum += (x[i].x + x[i].y) + (x[i].z + x[i].w);
    }
#pragma unroll
    for (int off = 16; off > 0; off >>= 1) sum += __shfl_xor(sum, off, 32);
    if ((tid & 31) == 0) redSum[tid >> 5] = sum;
    __syncthreads();
    float tot = 0.0f;
#pragma unroll
    for (int w = 0; w < 8; ++w) tot += redSum[w];
    const float inv = 1.0f / tot;
#pragma unroll
    for (int i = 0; i < 2; ++i) {
        x[i].x *= inv; x[i].y *= inv; x[i].z *= inv; x[i].w *= inv;
        p4[tid + i * 256] = x[i];
    }
}

// ---------------------------------------------------------------------------
// Kernel 4: combined = attn @ V -> bf16 merged-head layout [b*S+s][D].
// Grid: (S/64, 1, B*H); block 256.  K-chunk = 64 key positions.
// ---------------------------------------------------------------------------
__global__ __launch_bounds__(256) void av_kernel(const float* __restrict__ attn,
                                                 const __bf16* __restrict__ Vb,
                                                 __bf16* __restrict__ Cb) {
    const int bh = blockIdx.z;
    const int b = bh / H, h = bh % H;
    const int q0 = blockIdx.x * 64;

    __shared__ __bf16 lA[64 * 64];  // attn chunk [q][key]
    __shared__ __bf16 lV[64 * 64];  // V transposed: [dk][key]

    const int tid = threadIdx.x, lane = tid & 31, wave = tid >> 5;
    const float* Ah = attn + (size_t)bh * S * S;
    const unsigned short* Vu =
        (const unsigned short*)(Vb + (size_t)bh * S * DK);

    AccF acc[2];
#pragma unroll
    for (int t = 0; t < 2; ++t) acc[t].v = (v8f){0, 0, 0, 0, 0, 0, 0, 0};

    for (int p0 = 0; p0 < S; p0 += 64) {
        // attn chunk 64x64 fp32 -> bf16 via float4 loads
#pragma unroll
        for (int i = 0; i < 4; ++i) {
            const int idx = tid + i * 256;
            const int r = idx >> 4, c4 = idx & 15;
            const float4 xv =
                ((const float4*)(Ah + (size_t)(q0 + r) * S + p0))[c4];
            uint2 pv;
            pv.x = pk_bf16(xv.x, xv.y);
            pv.y = pk_bf16(xv.z, xv.w);
            ((uint2*)lA)[r * 16 + c4] = pv;
        }
        // V chunk 64(key)x64(dk) transposed into [dk][key]; pack key-pairs
#pragma unroll
        for (int i = 0; i < 8; ++i) {
            const int idx = tid + i * 256;  // 32 key-pairs x 64 dk
            const int kp = idx >> 6, nn = idx & 63;
            const size_t g0 = (size_t)(p0 + 2 * kp) * DK + nn;
            ((unsigned*)lV)[nn * 32 + kp] = pk_u16(Vu[g0], Vu[g0 + DK]);
        }
        __syncthreads();
#pragma unroll
        for (int t = 0; t < 2; ++t) {
            const int tile = wave * 2 + t;
            const int qt = tile >> 2, nt = tile & 3;
#pragma unroll
            for (int kk = 0; kk < 64; kk += 32) {
                const v16bf fa = load_frag_mk(lA + qt * 16 * 64 + kk, 64, lane);
                const v16bf fb = load_frag_mk(lV + nt * 16 * 64 + kk, 64, lane);
                acc[t].v = wmma_bf16(fa, fb, acc[t].v);
            }
        }
        __syncthreads();
    }

#pragma unroll
    for (int t = 0; t < 2; ++t) {
        const int tile = wave * 2 + t;
        const int qt = tile >> 2, nt = tile & 3;
        const int nl = lane & 15, hh = lane >> 4;
        const int dn = nt * 16 + nl;
#pragma unroll
        for (int i = 0; i < 8; ++i) {
            const int qq = q0 + qt * 16 + i + hh * 8;
            Cb[(size_t)(b * S + qq) * D + h * DK + dn] = (__bf16)acc[t].f[i];
        }
    }
}

// ---------------------------------------------------------------------------
// Kernel 5: out = combined * Wo + bo (fp32 out).  Grid: (M/64, D/64); block 256.
// A tile is bf16-resident -> async b128 global->LDS copies.
// ---------------------------------------------------------------------------
__global__ __launch_bounds__(256) void oproj_kernel(const __bf16* __restrict__ Cb,
                                                    const float* __restrict__ Wo,
                                                    const float* __restrict__ bo,
                                                    float* __restrict__ out) {
    __shared__ __bf16 lA[64 * 64];  // [m][k]
    __shared__ __bf16 lB[64 * 64];  // [n][k]

    const int tid = threadIdx.x, lane = tid & 31, wave = tid >> 5;
    const int m0 = blockIdx.x * 64;
    const int n0 = blockIdx.y * 64;

    AccF acc[2];
#pragma unroll
    for (int t = 0; t < 2; ++t) acc[t].v = (v8f){0, 0, 0, 0, 0, 0, 0, 0};

    for (int k0 = 0; k0 < D; k0 += 64) {
        // A: bf16 straight copy, async b128 (8 uint4 per 64-elem row)
#pragma unroll
        for (int i = 0; i < 2; ++i) {
            const int idx = tid + i * 256;
            const int r = idx >> 3, c = idx & 7;
            copy_b128_to_lds(&((uint4*)lA)[idx],
                             (const uint4*)(Cb + (size_t)(m0 + r) * D + k0) + c);
        }
        // B: Wo chunk transposed into [n][k]; pack k-pairs
#pragma unroll
        for (int i = 0; i < 8; ++i) {
            const int idx = tid + i * 256;
            const int kp = idx >> 6, nn = idx & 63;
            const size_t g0 = (size_t)(k0 + 2 * kp) * D + (n0 + nn);
            ((unsigned*)lB)[nn * 32 + kp] = pk_bf16(Wo[g0], Wo[g0 + D]);
            if (k0 + 64 < D)
                __builtin_prefetch(&Wo[g0 + (size_t)64 * D], 0, 1);
        }
        copy_fence();
        __syncthreads();
#pragma unroll
        for (int t = 0; t < 2; ++t) {
            const int tile = wave * 2 + t;
            const int qt = tile >> 2, nt = tile & 3;
#pragma unroll
            for (int kk = 0; kk < 64; kk += 32) {
                const v16bf fa = load_frag_mk(lA + qt * 16 * 64 + kk, 64, lane);
                const v16bf fb = load_frag_mk(lB + nt * 16 * 64 + kk, 64, lane);
                acc[t].v = wmma_bf16(fa, fb, acc[t].v);
            }
        }
        __syncthreads();
    }

#pragma unroll
    for (int t = 0; t < 2; ++t) {
        const int tile = wave * 2 + t;
        const int qt = tile >> 2, nt = tile & 3;
        const int nl = lane & 15, hh = lane >> 4;
        const int n  = n0 + nt * 16 + nl;
        const float bb = bo[n];
#pragma unroll
        for (int i = 0; i < 8; ++i) {
            const int m = m0 + qt * 16 + i + hh * 8;
            out[(size_t)m * D + n] = acc[t].f[i] + bb;
        }
    }
}

// ---------------------------------------------------------------------------
extern "C" void kernel_launch(void* const* d_in, const int* in_sizes, int n_in,
                              void* d_out, int out_size, void* d_ws, size_t ws_size,
                              hipStream_t stream) {
    (void)in_sizes; (void)n_in; (void)out_size; (void)ws_size;

    const float* query = (const float*)d_in[0];
    const float* key   = (const float*)d_in[1];
    const float* value = (const float*)d_in[2];
    const int*   mask  = (const int*)d_in[3];
    const float* Wq = (const float*)d_in[4];
    const float* bq = (const float*)d_in[5];
    const float* Wk = (const float*)d_in[6];
    const float* bk = (const float*)d_in[7];
    const float* Wv = (const float*)d_in[8];
    const float* bv = (const float*)d_in[9];
    const float* Wo = (const float*)d_in[10];
    const float* bo = (const float*)d_in[11];

    float* out  = (float*)d_out;             // [B,S,D]
    float* attn = out + (size_t)Bd * S * D;  // [B,H,S,S]

    const size_t elems = (size_t)Bd * S * D;  // 4,194,304
    __bf16* Qb = (__bf16*)d_ws;               // 8 MB each
    __bf16* Kb = Qb + elems;
    __bf16* Vb = Kb + elems;
    __bf16* Cb = Vb + elems;

    const int M = Bd * S;  // 4096

    qkv_proj_kernel<<<dim3(M / 64, D / 64, 3), 256, 0, stream>>>(
        query, key, value, Wq, bq, Wk, bk, Wv, bv, Qb, Kb, Vb);

    scores_kernel<<<dim3(S / 64, S / 64, Bd * H), 256, 0, stream>>>(Qb, Kb, mask,
                                                                    attn);

    softmax_kernel<<<dim3(Bd * H * S), 256, 0, stream>>>(attn);

    av_kernel<<<dim3(S / 64, 1, Bd * H), 256, 0, stream>>>(attn, Vb, Cb);

    oproj_kernel<<<dim3(M / 64, D / 64, 1), 256, 0, stream>>>(Cb, Wo, bo, out);
}
